// GE2Enet_36326833389809
// MI455X (gfx1250) — matrix operand order repaired and checked by reading, not compiled, not run on hardware
//
#include <hip/hip_runtime.h>

// GE2E loss on MI455X (gfx1250, wave32, WMMA).
// S=1024 speakers, U=64 utterances, D=256. Dominant cost: 65536x1024x256 GEMM
// (34.4 GFLOP) done with v_wmma_f32_16x16x32_f16 (f16 in, f32 accumulate),
// fused with online sum-exp (logits bounded in [-15,5] so no running max needed).
// Column-tile loop unrolled x2 with two independent WMMA accumulator chains to
// keep the XDL pipe busy (single-chain WMMAs serialize on the C-operand RAW).

typedef __attribute__((ext_vector_type(16))) _Float16 v16h;
typedef __attribute__((ext_vector_type(8)))  _Float16 v8h;
typedef __attribute__((ext_vector_type(8)))  float    v8f;

#define S_SPK 1024
#define U_UTT 64
#define D_EMB 256
#define N_ROWS (S_SPK * U_UTT)   // 65536
#define EPS 1e-5f

// ---------------------------------------------------------------------------
// Block-wide 3-value reduction (256 threads = 8 waves, wave32).
// Fixed summation order -> deterministic.
// ---------------------------------------------------------------------------
__device__ __forceinline__ float3 blockReduce3(float a, float b, float c, float* red) {
#pragma unroll
  for (int off = 16; off > 0; off >>= 1) {
    a += __shfl_xor(a, off);
    b += __shfl_xor(b, off);
    c += __shfl_xor(c, off);
  }
  const int wid = threadIdx.x >> 5;
  __syncthreads();  // protect `red` from previous round's readers
  if ((threadIdx.x & 31) == 0) {
    red[wid * 4 + 0] = a;
    red[wid * 4 + 1] = b;
    red[wid * 4 + 2] = c;
  }
  __syncthreads();
  a = 0.f; b = 0.f; c = 0.f;
#pragma unroll
  for (int w = 0; w < 8; ++w) {
    a += red[w * 4 + 0];
    b += red[w * 4 + 1];
    c += red[w * 4 + 2];
  }
  return make_float3(a, b, c);
}

// ---------------------------------------------------------------------------
// Kernel A: per speaker -> normalized embeds (f16), inclusive centroid (f16),
// exclusive-centroid similarity (f32). One block per speaker, thread = dim d.
// ---------------------------------------------------------------------------
__global__ __launch_bounds__(256) void ge2e_prep(const float* __restrict__ emb,
                                                 _Float16* __restrict__ An,
                                                 _Float16* __restrict__ Bc,
                                                 float* __restrict__ simd) {
  __shared__ float red[8 * 4];
  const int s = blockIdx.x;
  const int t = threadIdx.x;  // dimension index d
  const float* base = emb + (size_t)s * U_UTT * D_EMB + t;

  float colsum = 0.f;
#pragma unroll 4
  for (int u = 0; u < U_UTT; ++u) colsum += base[(size_t)u * D_EMB];

  // inclusive centroid, L2-normalized with +eps
  const float mean = colsum * (1.0f / (float)U_UTT);
  float3 rr = blockReduce3(mean * mean, 0.f, 0.f, red);
  const float cin = mean / (sqrtf(rr.x) + EPS);
  Bc[(size_t)s * D_EMB + t] = (_Float16)cin;

  for (int u = 0; u < U_UTT; ++u) {
    const float e = base[(size_t)u * D_EMB];
    const float craw = (colsum - e) * (1.0f / (float)(U_UTT - 1));
    // a = ||e||^2, b = e.colsum, c = ||c_excl_raw||^2  (one reduction round)
    float3 q = blockReduce3(e * e, e * colsum, craw * craw, red);
    const float ne = sqrtf(q.x) + EPS;
    const float nc = sqrtf(q.z) + EPS;
    // e . c_raw = (e.colsum - ||e||^2)/(U-1)
    const float dot = (q.y - q.x) * (1.0f / (float)(U_UTT - 1));
    const float sim = dot / (ne * nc);
    An[((size_t)s * U_UTT + u) * D_EMB + t] = (_Float16)(e / ne);
    if (t == 0) simd[s * U_UTT + u] = sim;
  }
}

// ---------------------------------------------------------------------------
// WMMA operand loaders (wave32 f16 16x16x32 layouts).
// A (16xK, 16-bit): lanes 0-15 & 16-31 both hold row M = lane%16;
//   VGPRs 0-3: K {0..7 | 8..15}, VGPRs 4-7: K {16..23 | 24..31}.
// B (Kx16, 16-bit): lane = column N; same K order as A:
//   VGPRs 0-3: K {0..7 | 8..15}, VGPRs 4-7: K {16..23 | 24..31}.
// Both source buffers are K-contiguous, so these are plain b128 loads.
// ---------------------------------------------------------------------------
__device__ __forceinline__ v16h loadA_k32(const _Float16* rowbase, int laneHi) {
  const _Float16* p = rowbase + (laneHi ? 8 : 0);
  v8h lo = *(const v8h*)(p);        // K {0..7}  or {8..15}
  v8h hi = *(const v8h*)(p + 16);   // K {16..23} or {24..31}
  v16h r;
#pragma unroll
  for (int i = 0; i < 8; ++i) { r[i] = lo[i]; r[8 + i] = hi[i]; }
  return r;
}

__device__ __forceinline__ v16h loadB_k32(const _Float16* colbase, int laneHi) {
  const _Float16* p = colbase + (laneHi ? 8 : 0);
  v8h lo = *(const v8h*)(p);
  v8h hi = *(const v8h*)(p + 16);
  v16h r;
#pragma unroll
  for (int i = 0; i < 8; ++i) { r[i] = lo[i]; r[8 + i] = hi[i]; }
  return r;
}

// ---------------------------------------------------------------------------
// Kernel B: fused GEMM + diagonal substitution + sum-exp.
// 8 waves/block, each wave owns 16 consecutive rows (one speaker quarter),
// sweeps all 64 column tiles (2 per iteration); A held in registers.
// ---------------------------------------------------------------------------
__global__ __launch_bounds__(256) void ge2e_gemm(const _Float16* __restrict__ An,
                                                 const _Float16* __restrict__ Bc,
                                                 const float* __restrict__ simd,
                                                 const float* __restrict__ wp,
                                                 const float* __restrict__ bp,
                                                 float* __restrict__ rowloss) {
  const int lane   = threadIdx.x & 31;
  const int wave   = threadIdx.x >> 5;
  const int mtile  = blockIdx.x * 8 + wave;   // 0..4095
  const int mbase  = mtile * 16;
  const int laneHi = lane >> 4;
  const int lane16 = lane & 15;
  const int arow   = mbase + lane16;

  const float w    = wp[0];
  const float bias = bp[0];

  // 16 consecutive rows lie in one speaker (64 rows/speaker, tiles of 16)
  const int sTile    = mbase >> 6;     // target column for every row here
  const int jtDiag   = sTile >> 4;     // column tile containing the diagonal
  const int laneDiag = sTile & 15;     // lane within that tile

  // Keep the whole A row-tile (16 rows x K=256) in registers: 8 x v16h
  v16h Areg[8];
#pragma unroll
  for (int kc = 0; kc < 8; ++kc)
    Areg[kc] = loadA_k32(An + (size_t)arow * D_EMB + kc * 32, laneHi);

  float rowsum[8];
#pragma unroll
  for (int r = 0; r < 8; ++r) rowsum[r] = 0.f;

  for (int jt = 0; jt < 64; jt += 2) {
    // Two column tiles per iteration -> two independent WMMA chains.
    const int spk0 = jt * 16 + lane16;
    const _Float16* bb0 = Bc + (size_t)spk0 * D_EMB;
    const _Float16* bb1 = bb0 + 16 * D_EMB;

    v8f acc0 = {};
    v8f acc1 = {};
#pragma unroll
    for (int kc = 0; kc < 8; ++kc) {
      v16h b0 = loadB_k32(bb0 + kc * 32, laneHi);
      v16h b1 = loadB_k32(bb1 + kc * 32, laneHi);
      acc0 = __builtin_amdgcn_wmma_f32_16x16x32_f16(
          false, Areg[kc], false, b0, (short)0, acc0, false, false);
      acc1 = __builtin_amdgcn_wmma_f32_16x16x32_f16(
          false, Areg[kc], false, b1, (short)0, acc1, false, false);
    }

#pragma unroll
    for (int half = 0; half < 2; ++half) {
      const bool diagTile = ((jt + half) == jtDiag);
#pragma unroll
      for (int r = 0; r < 8; ++r) {
        float t  = (half ? acc1[r] : acc0[r]) * w + bias;  // logit in [-15, 5]
        float ev = __expf(t);
        if (diagTile && lane16 == laneDiag) ev = 0.f;  // drop diag, replaced later
        ev += __shfl_xor(ev, 1);
        ev += __shfl_xor(ev, 2);
        ev += __shfl_xor(ev, 4);
        ev += __shfl_xor(ev, 8);           // reduce within each 16-lane half
        rowsum[r] += ev;
      }
    }
  }

  // Finalize: add exclusive-centroid diagonal term; loss_r = LSE - logit_target
#pragma unroll
  for (int r = 0; r < 8; ++r) {
    const int grow = mbase + r + laneHi * 8;
    const float tt = simd[grow] * w + bias;
    const float total = rowsum[r] + __expf(tt);
    const float l = logf(total) - tt;
    if (lane16 == 0) rowloss[grow] = l;    // lanes 0 and 16 cover the 16 rows
  }
}

// ---------------------------------------------------------------------------
// Kernel C: deterministic final mean over 65536 per-row losses.
// ---------------------------------------------------------------------------
__global__ __launch_bounds__(256) void ge2e_reduce(const float* __restrict__ rowloss,
                                                   float* __restrict__ out) {
  __shared__ float red[8];
  float s = 0.f;
  for (int i = threadIdx.x; i < N_ROWS; i += 256) s += rowloss[i];
#pragma unroll
  for (int off = 16; off > 0; off >>= 1) s += __shfl_xor(s, off);
  if ((threadIdx.x & 31) == 0) red[threadIdx.x >> 5] = s;
  __syncthreads();
  if (threadIdx.x == 0) {
    float tot = 0.f;
#pragma unroll
    for (int wv = 0; wv < 8; ++wv) tot += red[wv];
    out[0] = tot * (1.0f / (float)N_ROWS);
  }
}

// ---------------------------------------------------------------------------
// Launch. Workspace layout (33 MB):
//   [0, 32MB)            An   : normalized embeds, f16, [65536 x 256]
//   [32MB, +512KB)       Bc   : inclusive centroids, f16, [1024 x 256]
//   [+256KB)             simd : exclusive-centroid similarity, f32 [65536]
//   [+256KB)             rowloss : per-row loss, f32 [65536]
// ---------------------------------------------------------------------------
extern "C" void kernel_launch(void* const* d_in, const int* in_sizes, int n_in,
                              void* d_out, int out_size, void* d_ws, size_t ws_size,
                              hipStream_t stream) {
  const float* emb = (const float*)d_in[0];   // [S*U, D] fp32
  const float* w   = (const float*)d_in[1];   // similarity_weight [1]
  const float* b   = (const float*)d_in[2];   // similarity_bias   [1]
  // d_in[3] / d_in[4] are S and U scalars; baked in as compile-time constants.

  char* ws = (char*)d_ws;
  const size_t offAn = 0;
  const size_t offBc = (size_t)N_ROWS * D_EMB * sizeof(_Float16);          // 32 MB
  const size_t offSd = offBc + (size_t)S_SPK * D_EMB * sizeof(_Float16);   // +512 KB
  const size_t offRl = offSd + (size_t)N_ROWS * sizeof(float);             // +256 KB

  _Float16* An      = (_Float16*)(ws + offAn);
  _Float16* Bc      = (_Float16*)(ws + offBc);
  float*    simd    = (float*)(ws + offSd);
  float*    rowloss = (float*)(ws + offRl);

  ge2e_prep<<<S_SPK, 256, 0, stream>>>(emb, An, Bc, simd);
  ge2e_gemm<<<N_ROWS / 128, 256, 0, stream>>>(An, Bc, simd, w, b, rowloss);
  ge2e_reduce<<<1, 256, 0, stream>>>(rowloss, (float*)d_out);
}
